// HGT_5583457485561
// MI455X (gfx1250) — matrix operand order repaired and checked by reading, not compiled, not run on hardware
//
#include <hip/hip_runtime.h>
#include <hip/hip_bf16.h>
#include <cstdint>

// ---------------------------------------------------------------------------
// HGT forward for MI455X (gfx1250, wave32).
// Strategy: all dense GEMMs via V_WMMA_F32_16X16X32_BF16 (bf16 in, f32 acc).
// Per-edge 32x32 relation transforms hoisted to per-node block-diagonal GEMMs.
// Segment softmax via ordered-uint atomicMax + f32 atomicAdd scatter.
// ---------------------------------------------------------------------------

typedef __attribute__((ext_vector_type(16))) __bf16 v16bf;
typedef __attribute__((ext_vector_type(8)))  float  v8f;

#define DEVINL __device__ __forceinline__

static const int NM = 50000, ND = 10000, NA = 25000, NTOT = 85000;
static const int FIN = 1024, HID = 256, HEADS = 8, HD = 32, NOUT = 4;

// Native fptrunc: gfx1250 lowers this to v_cvt_pk_bf16_f32 (RNE), 2 floats/op.
DEVINL __bf16 f2bf(float f) { return (__bf16)f; }

DEVINL unsigned f2ord(float f) {   // order-preserving float->uint
  unsigned u = __float_as_uint(f);
  return (u >> 31) ? ~u : (u | 0x80000000u);
}
DEVINL float ord2f(unsigned u) {
  unsigned b = (u & 0x80000000u) ? (u & 0x7FFFFFFFu) : ~u;
  return __uint_as_float(b);
}
DEVINL float gelu_tanh(float x) {  // jax.nn.gelu approximate=True
  float x3 = x * x * x;
  return 0.5f * x * (1.0f + tanhf(0.7978845608028654f * (x + 0.044715f * x3)));
}

DEVINL v16bf cvt16(float4 a0, float4 a1, float4 a2, float4 a3) {
  v16bf av;
  av[0] = f2bf(a0.x);  av[1] = f2bf(a0.y);  av[2] = f2bf(a0.z);  av[3] = f2bf(a0.w);
  av[4] = f2bf(a1.x);  av[5] = f2bf(a1.y);  av[6] = f2bf(a1.z);  av[7] = f2bf(a1.w);
  av[8] = f2bf(a2.x);  av[9] = f2bf(a2.y);  av[10] = f2bf(a2.z); av[11] = f2bf(a2.w);
  av[12] = f2bf(a3.x); av[13] = f2bf(a3.y); av[14] = f2bf(a3.z); av[15] = f2bf(a3.w);
  return av;
}

// --------------------------- weight prep -----------------------------------
// W[K,N] f32 (row major) -> Bt[N,K] bf16 (transposed, contiguous K per column)
__global__ void k_wt(const float* __restrict__ W, __bf16* __restrict__ Bt,
                     int K, int N) {
  int i = blockIdx.x * blockDim.x + threadIdx.x;
  if (i >= K * N) return;
  int k = i / N, n = i - k * N;
  Bt[(size_t)n * K + k] = f2bf(W[i]);
}

// a[8][32][32] f32 -> block-diagonal Bt[256(n)][256(k)] bf16 (transposed)
// BD[k=h*32+d][n=h*32+f] = a[h][d][f]
__global__ void k_bd(const float* __restrict__ a, __bf16* __restrict__ Bt) {
  int i = blockIdx.x * blockDim.x + threadIdx.x;
  if (i >= HID * HID) return;
  int n = i >> 8, k = i & 255;
  float v = ((n >> 5) == (k >> 5))
                ? a[(n >> 5) * (HD * HD) + (k & 31) * HD + (n & 31)]
                : 0.0f;
  Bt[i] = f2bf(v);
}

__global__ void k_fill(unsigned* __restrict__ p, unsigned v, unsigned n) {
  unsigned i = blockIdx.x * blockDim.x + threadIdx.x;
  if (i < n) p[i] = v;
}

// --------------------------- WMMA GEMM -------------------------------------
// C[M,N] = A[M,K](f32, cvt->bf16) @ Bt[N,K](bf16 pre-transposed) + bias
// epi==1: C = beta*gelu(acc+bias) + (1-beta)*skipsrc   (beta = sigmoid(*skipv))
// One wave computes one 16x16 tile; 8 waves/block along M; grid.y tiles N.
// - A-row out-of-range: clamp the pointer (garbage feeds only D rows never
//   stored) -> no divergent branch, EXEC all-1s for WMMA.
// - Software pipeline: loads for iteration i+1 issue before wmma_i, so the
//   pre-wmma wait is long satisfied and regalloc double-buffers A/B (no WAR
//   nops against the wmma's source registers).
__global__ void k_gemm(const float* __restrict__ A, const __bf16* __restrict__ Bt,
                       const float* __restrict__ bias, float* __restrict__ C,
                       int M, int N, int K, int epi,
                       const float* __restrict__ skipv,
                       const float* __restrict__ skipsrc) {
  const int lane = threadIdx.x & 31;
  const int wave = threadIdx.x >> 5;
  const int mt = blockIdx.x * (blockDim.x >> 5) + wave;
  const int m0 = mt * 16;
  if (m0 >= M) return;                       // wave-uniform: EXEC stays all-1s
  const int half = lane >> 4;                // lane-half selects K sub-range
  const int lm = lane & 15;
  int row = m0 + lm;                         // A row this lane feeds
  row = row < M ? row : M - 1;               // clamp (see note above)
  const int col = blockIdx.y * 16 + lm;      // C column this lane owns
  // A: lanes 0-15 hold K..K+15, lanes 16-31 hold K+16..K+31 (contiguous)
  const float* __restrict__ arow = A + (size_t)row * K + 16 * half;
  // B: lanes 0-15 hold K..K+15, lanes 16-31 hold K+16..K+31 (contiguous 32B)
  const __bf16* __restrict__ brow = Bt + (size_t)col * K + 16 * half;

  v8f acc;
#pragma unroll
  for (int r = 0; r < 8; ++r) acc[r] = 0.0f;

  // ---- software-pipelined K loop (K is a multiple of 32, K >= 32) ----
  float4 ca0 = *(const float4*)(arow);
  float4 ca1 = *(const float4*)(arow + 4);
  float4 ca2 = *(const float4*)(arow + 8);
  float4 ca3 = *(const float4*)(arow + 12);
  v16bf cb = *(const v16bf*)(brow);
#pragma unroll 4
  for (int k = 32; k < K; k += 32) {
    const float4 na0 = *(const float4*)(arow + k);
    const float4 na1 = *(const float4*)(arow + k + 4);
    const float4 na2 = *(const float4*)(arow + k + 8);
    const float4 na3 = *(const float4*)(arow + k + 12);
    const v16bf nb = *(const v16bf*)(brow + k);
    const v16bf av = cvt16(ca0, ca1, ca2, ca3);
    acc = __builtin_amdgcn_wmma_f32_16x16x32_bf16(
        false, av, false, cb, (short)0, acc, false, false);
    ca0 = na0; ca1 = na1; ca2 = na2; ca3 = na3; cb = nb;
  }
  {
    const v16bf av = cvt16(ca0, ca1, ca2, ca3);
    acc = __builtin_amdgcn_wmma_f32_16x16x32_bf16(
        false, av, false, cb, (short)0, acc, false, false);
  }

  const float bb = bias ? bias[col] : 0.0f;
  if (epi == 0) {
#pragma unroll
    for (int r = 0; r < 8; ++r) {
      int m = m0 + r + 8 * half;             // D: VGPR r -> row r + 8*half
      if (m < M) C[(size_t)m * N + col] = acc[r] + bb;
    }
  } else {
    const float beta = 1.0f / (1.0f + __expf(-skipv[0]));
#pragma unroll
    for (int r = 0; r < 8; ++r) {
      int m = m0 + r + 8 * half;
      if (m < M) {
        float x = acc[r] + bb;
        C[(size_t)m * N + col] =
            beta * gelu_tanh(x) + (1.0f - beta) * skipsrc[(size_t)m * N + col];
      }
    }
  }
}

// --------------------------- edge kernels ----------------------------------
// phase 1: score = (q[dst] . ka[src]) * prio[h] / sqrt(HD); atomicMax into amax
__global__ void k_edge_score(const int* __restrict__ si, const int* __restrict__ di,
                             const float* __restrict__ qdst,
                             const float* __restrict__ ka,
                             const float* __restrict__ prio,
                             float* __restrict__ al,
                             unsigned* __restrict__ amax, int E) {
  int t = blockIdx.x * blockDim.x + threadIdx.x;
  if (t >= E * HEADS) return;
  int e = t >> 3, h = t & 7;
  int s = si[e], d = di[e];
  const float* qp = qdst + (size_t)d * HID + h * HD;
  const float* kp = ka + (size_t)s * HID + h * HD;
  float acc = 0.0f;
#pragma unroll
  for (int f = 0; f < HD; ++f) acc += qp[f] * kp[f];
  float sc = acc * prio[h] * 0.17677669529663688f;  // 1/sqrt(32)
  al[t] = sc;
  atomicMax(amax + (size_t)d * HEADS + h, f2ord(sc));
}

// phase 2: ex = exp(al - max); den += ex; agg += ex * vm[src]
__global__ void k_edge_agg(const int* __restrict__ si, const int* __restrict__ di,
                           const float* __restrict__ al,
                           const unsigned* __restrict__ amax,
                           float* __restrict__ den,
                           const float* __restrict__ vm,
                           float* __restrict__ agg, int E) {
  int t = blockIdx.x * blockDim.x + threadIdx.x;
  if (t >= E * HEADS) return;
  int e = t >> 3, h = t & 7;
  int s = si[e], d = di[e];
  float mx = ord2f(amax[(size_t)d * HEADS + h]);
  float ex = __expf(al[t] - mx);
  atomicAdd(den + (size_t)d * HEADS + h, ex);
  const float* vp = vm + (size_t)s * HID + h * HD;
  float* ap = agg + (size_t)d * HID + h * HD;
#pragma unroll
  for (int f = 0; f < HD; ++f) atomicAdd(ap + f, ex * vp[f]);
}

// agg /= den (guard den>0). i = node*256 + head*32 + f  ->  i>>5 = node*8+head
__global__ void k_div(float* __restrict__ agg, const float* __restrict__ den,
                      int n) {
  int i = blockIdx.x * blockDim.x + threadIdx.x;
  if (i >= n) return;
  float d = den[i >> 5];
  agg[i] = agg[i] / (d > 0.0f ? d : 1.0f);
}

// final classifier: Y[M,4] = X[M,256] @ W[256,4] + b
__global__ void k_head(const float* __restrict__ X, const float* __restrict__ W,
                       const float* __restrict__ b, float* __restrict__ Y,
                       int M) {
  int t = blockIdx.x * blockDim.x + threadIdx.x;
  if (t >= M * NOUT) return;
  int m = t >> 2, o = t & 3;
  float acc = b[o];
  const float* xr = X + (size_t)m * HID;
  for (int c = 0; c < HID; ++c) acc += xr[c] * W[c * NOUT + o];
  Y[t] = acc;
}

// --------------------------- host orchestration ----------------------------
extern "C" void kernel_launch(void* const* d_in, const int* in_sizes, int n_in,
                              void* d_out, int out_size, void* d_ws,
                              size_t ws_size, hipStream_t stream) {
  (void)in_sizes; (void)n_in; (void)out_size; (void)ws_size;
  const int typeN[3] = {NM, ND, NA};
  const int typeOff[3] = {0, NM, NM + ND};
  struct RelMeta { int src, dst, E; };
  const RelMeta rels[4] = {{0, 1, 150000}, {1, 0, 150000},
                           {0, 2, 300000}, {2, 0, 300000}};

  char* ws = (char*)d_ws;
  size_t off = 0;
  auto alloc = [&](size_t bytes) -> char* {
    char* p = ws + off;
    off = (off + bytes + 255) & ~(size_t)255;
    return p;
  };

  __bf16 *wtin[3], *wtk[3], *wtq[3], *wtv[3], *wta[3];
  for (int t = 0; t < 3; ++t) wtin[t] = (__bf16*)alloc((size_t)FIN * HID * 2);
  for (int t = 0; t < 3; ++t) wtk[t] = (__bf16*)alloc((size_t)HID * HID * 2);
  for (int t = 0; t < 3; ++t) wtq[t] = (__bf16*)alloc((size_t)HID * HID * 2);
  for (int t = 0; t < 3; ++t) wtv[t] = (__bf16*)alloc((size_t)HID * HID * 2);
  for (int t = 0; t < 3; ++t) wta[t] = (__bf16*)alloc((size_t)HID * HID * 2);
  __bf16 *bda[4], *bdm[4];
  for (int r = 0; r < 4; ++r) bda[r] = (__bf16*)alloc((size_t)HID * HID * 2);
  for (int r = 0; r < 4; ++r) bdm[r] = (__bf16*)alloc((size_t)HID * HID * 2);

  float* hb = (float*)alloc((size_t)NTOT * HID * 4);
  float* kb = (float*)alloc((size_t)NTOT * HID * 4);  // reused as agg later
  float* qb = (float*)alloc((size_t)NTOT * HID * 4);
  float* vb = (float*)alloc((size_t)NTOT * HID * 4);  // reused as out later
  float *ka[4], *vmr[4], *al[4];
  for (int r = 0; r < 4; ++r)
    ka[r] = (float*)alloc((size_t)typeN[rels[r].src] * HID * 4);
  for (int r = 0; r < 4; ++r)
    vmr[r] = (float*)alloc((size_t)typeN[rels[r].src] * HID * 4);
  for (int r = 0; r < 4; ++r)
    al[r] = (float*)alloc((size_t)rels[r].E * HEADS * 4);
  unsigned* amax = (unsigned*)alloc((size_t)NTOT * HEADS * 4);
  float* den = (float*)alloc((size_t)NTOT * HEADS * 4);
  float* agg = kb;   // k dead after block-diag GEMMs
  float* outb = vb;  // v dead after block-diag GEMMs

  auto F = [&](int i) { return (const float*)d_in[i]; };
  auto I = [&](int i) { return (const int*)d_in[i]; };

  // 1. convert weights to transposed bf16 (+ block-diagonal relation mats)
  for (int t = 0; t < 3; ++t) {
    int base = 23 + t * 11;
    int nin = FIN * HID, nh = HID * HID;
    k_wt<<<(nin + 255) / 256, 256, 0, stream>>>(F(base + 0), wtin[t], FIN, HID);
    k_wt<<<(nh + 255) / 256, 256, 0, stream>>>(F(base + 2), wtk[t], HID, HID);
    k_wt<<<(nh + 255) / 256, 256, 0, stream>>>(F(base + 4), wtq[t], HID, HID);
    k_wt<<<(nh + 255) / 256, 256, 0, stream>>>(F(base + 6), wtv[t], HID, HID);
    k_wt<<<(nh + 255) / 256, 256, 0, stream>>>(F(base + 8), wta[t], HID, HID);
  }
  for (int r = 0; r < 4; ++r) {
    int base = 3 + r * 5;
    k_bd<<<(HID * HID + 255) / 256, 256, 0, stream>>>(F(base + 2), bda[r]);
    k_bd<<<(HID * HID + 255) / 256, 256, 0, stream>>>(F(base + 3), bdm[r]);
  }

  auto gemm = [&](const float* A, const __bf16* Bt, const float* bias, float* C,
                  int M, int K, int epi, const float* skipv,
                  const float* skips) {
    dim3 grid((unsigned)(((M + 15) / 16 + 7) / 8), HID / 16);
    k_gemm<<<grid, 256, 0, stream>>>(A, Bt, bias, C, M, HID, K, epi, skipv,
                                     skips);
  };

  // 2. input projection h = x @ W_in + b_in
  for (int t = 0; t < 3; ++t)
    gemm(F(t), wtin[t], F(23 + t * 11 + 1), hb + (size_t)typeOff[t] * HID,
         typeN[t], FIN, 0, nullptr, nullptr);

  // 3. k/q/v projections
  for (int t = 0; t < 3; ++t) {
    int base = 23 + t * 11;
    size_t o = (size_t)typeOff[t] * HID;
    gemm(hb + o, wtk[t], F(base + 3), kb + o, typeN[t], HID, 0, nullptr, nullptr);
    gemm(hb + o, wtq[t], F(base + 5), qb + o, typeN[t], HID, 0, nullptr, nullptr);
    gemm(hb + o, wtv[t], F(base + 7), vb + o, typeN[t], HID, 0, nullptr, nullptr);
  }

  // 4. per-relation hoisted transforms: ka = k @ BD(a), vm = v @ BD(m)
  for (int r = 0; r < 4; ++r) {
    size_t so = (size_t)typeOff[rels[r].src] * HID;
    int M = typeN[rels[r].src];
    gemm(kb + so, bda[r], nullptr, ka[r], M, HID, 0, nullptr, nullptr);
    gemm(vb + so, bdm[r], nullptr, vmr[r], M, HID, 0, nullptr, nullptr);
  }

  // 5. init softmax stats + agg (agg overlays kb, dead from here on)
  {
    unsigned n1 = (unsigned)(NTOT * HEADS);
    unsigned n2 = (unsigned)(NTOT * HID);
    k_fill<<<(n1 + 255) / 256, 256, 0, stream>>>(amax, 0x007FFFFFu, n1); // ord(-inf)
    k_fill<<<(n1 + 255) / 256, 256, 0, stream>>>((unsigned*)den, 0u, n1);
    k_fill<<<(n2 + 255) / 256, 256, 0, stream>>>((unsigned*)agg, 0u, n2);
  }

  // 6. edge scores + segment max
  for (int r = 0; r < 4; ++r) {
    int base = 3 + r * 5, E = rels[r].E;
    size_t dOff = (size_t)typeOff[rels[r].dst];
    int nt = E * HEADS;
    k_edge_score<<<(nt + 255) / 256, 256, 0, stream>>>(
        I(base), I(base + 1), qb + dOff * HID, ka[r], F(base + 4), al[r],
        amax + dOff * HEADS, E);
  }
  // 7. exp + segment sum + weighted scatter-aggregate
  for (int r = 0; r < 4; ++r) {
    int base = 3 + r * 5, E = rels[r].E;
    size_t dOff = (size_t)typeOff[rels[r].dst];
    int nt = E * HEADS;
    k_edge_agg<<<(nt + 255) / 256, 256, 0, stream>>>(
        I(base), I(base + 1), al[r], amax + dOff * HEADS, den + dOff * HEADS,
        vmr[r], agg + dOff * HID, E);
  }

  // 8. normalize
  {
    int n = NTOT * HID;
    k_div<<<(n + 255) / 256, 256, 0, stream>>>(agg, den, n);
  }

  // 9. output projection + GELU + gated skip (out overlays vb)
  for (int t = 0; t < 3; ++t) {
    int base = 23 + t * 11;
    size_t o = (size_t)typeOff[t] * HID;
    gemm(agg + o, wta[t], F(base + 9), outb + o, typeN[t], HID, 1,
         F(base + 10), hb + o);
  }

  // 10. classifier on movie nodes
  {
    int nt = NM * NOUT;
    k_head<<<(nt + 255) / 256, 256, 0, stream>>>(outb, F(56), F(57),
                                                 (float*)d_out, NM);
  }
}
